// ScaleCausalAttention_41936060678369
// MI455X (gfx1250) — compile-verified, matrix operand order checked by reading, not run on hardware
//
#include <hip/hip_runtime.h>
#include <hip/hip_bf16.h>

// ---------------------------------------------------------------------------
// ScaleCausalAttention for MI455X (gfx1250): bf16 WMMA pipeline.
//   B=64, N=341 (pad 352), D=768, H=12, hd=64.
//   GEMMs: double-buffered LDS staging via global_load_async_to_lds_b128
//          (ASYNCcnt) feeding v_wmma_f32_16x16x32_bf16.
//   Attention: flash-style, online softmax, P restaged through LDS.
// ---------------------------------------------------------------------------

typedef __bf16 bf16;
typedef __attribute__((ext_vector_type(16))) __bf16 v16bf;
typedef __attribute__((ext_vector_type(8)))  __bf16 v8bf;
typedef __attribute__((ext_vector_type(8)))  float  v8f;

#define WMMA_BF16(a, b, c) \
  __builtin_amdgcn_wmma_f32_16x16x32_bf16(false, (a), false, (b), (short)0, (c), false, false)

#define B_   64
#define N_   341
#define NPAD 352
#define D_   768
#define H_   12
#define HD_  64
#define M_   (B_ * N_)      // 21824 = 341 * 64
#define E3_  (3 * D_)       // 2304
#define KSTEPS (D_ / 32)    // 24

// ---------------------------------------------------------------------------
// 16-bit WMMA A/B fragment from a row-major 32-wide row (global or LDS).
// Lane (g = lane>>4, r = lane&15):
//   elements 0..7  -> K = kb + 8*g + 0..7
//   elements 8..15 -> K = kb + 16 + 8*g + 0..7
// ---------------------------------------------------------------------------
static __device__ __forceinline__ v16bf load_frag(const bf16* rowp, int kb, int g) {
  v8bf lo = *reinterpret_cast<const v8bf*>(rowp + kb + 8 * g);
  v8bf hi = *reinterpret_cast<const v8bf*>(rowp + kb + 16 + 8 * g);
  return __builtin_shufflevector(lo, hi, 0, 1, 2, 3, 4, 5, 6, 7,
                                 8, 9, 10, 11, 12, 13, 14, 15);
}

// Async 16-byte global -> LDS copy (per lane). Tracked by ASYNCcnt.
static __device__ __forceinline__ void async_copy_b128(void* lds_dst, const bf16* gsrc) {
  unsigned lds_off = (unsigned)(size_t)lds_dst;          // low 32b of LDS aperture ptr
  unsigned long long gaddr = (unsigned long long)(size_t)gsrc;
  asm volatile("global_load_async_to_lds_b128 %0, %1, off"
               :: "v"(lds_off), "v"(gaddr) : "memory");
}

static __device__ __forceinline__ int scale_of(int t) {
  // token t -> pyramid scale id (CLS=0, then 4/16/64/256 tokens)
  return (t == 0) ? 0 : (t < 5) ? 1 : (t < 21) ? 2 : (t < 85) ? 3 : 4;
}

// ---------------------------------------------------------------------------
// Kernel 1: f32 -> bf16 downcast (grid-stride).
// ---------------------------------------------------------------------------
__global__ void cvt_f32_bf16_kernel(const float* __restrict__ src,
                                    bf16* __restrict__ dst, int n) {
  int i = blockIdx.x * blockDim.x + threadIdx.x;
  int stride = gridDim.x * blockDim.x;
  for (; i < n; i += stride) dst[i] = (bf16)src[i];
}

// ---------------------------------------------------------------------------
// Shared GEMM tile pipeline: acc[4] += A(64x768 rows m0blk..) @ W^T(rows e0..)
// Double-buffered LDS staging; 4 waves, each owns 16(M) x 64(E) of the tile.
// ---------------------------------------------------------------------------
static __device__ __forceinline__ void gemm_tile_pipeline(
    const bf16* __restrict__ aG,   // &A[m0blk * D_]
    const bf16* __restrict__ wG,   // &W[e0 * D_]
    bf16 (*Asm)[64][32], bf16 (*Wsm)[64][32],
    int tid, int wid, int g, int cl, v8f acc[4]) {
  // Per-thread copy slice: chunks c and c+128 of the 256 16B chunks per tile.
  const int r0 = tid >> 2, cc0 = (tid & 3) * 8;          // chunk c  = tid
  const int r1 = (tid + 128) >> 2, cc1 = cc0;            // chunk c+128

  auto issue = [&](int buf, int kb) {
    async_copy_b128(&Asm[buf][r0][cc0], aG + (size_t)r0 * D_ + kb + cc0);
    async_copy_b128(&Asm[buf][r1][cc1], aG + (size_t)r1 * D_ + kb + cc1);
    async_copy_b128(&Wsm[buf][r0][cc0], wG + (size_t)r0 * D_ + kb + cc0);
    async_copy_b128(&Wsm[buf][r1][cc1], wG + (size_t)r1 * D_ + kb + cc1);
  };

  issue(0, 0);
  for (int ks = 0; ks < KSTEPS; ++ks) {
    const int cur = ks & 1;
    if (ks + 1 < KSTEPS) {
      issue(cur ^ 1, (ks + 1) * 32);
      asm volatile("s_wait_asynccnt 0x4" ::: "memory");  // this wave's prev 4 done
    } else {
      asm volatile("s_wait_asynccnt 0x0" ::: "memory");
    }
    __syncthreads();                                     // all waves' copies landed

    const v16bf af = load_frag(&Asm[cur][wid * 16 + cl][0], 0, g);
    acc[0] = WMMA_BF16(af, load_frag(&Wsm[cur][0 * 16 + cl][0], 0, g), acc[0]);
    acc[1] = WMMA_BF16(af, load_frag(&Wsm[cur][1 * 16 + cl][0], 0, g), acc[1]);
    acc[2] = WMMA_BF16(af, load_frag(&Wsm[cur][2 * 16 + cl][0], 0, g), acc[2]);
    acc[3] = WMMA_BF16(af, load_frag(&Wsm[cur][3 * 16 + cl][0], 0, g), acc[3]);

    __syncthreads();                                     // buffer free for reuse
  }
}

// ---------------------------------------------------------------------------
// Kernel 2: QKV GEMM.  qkv[m,e] = sum_k x[m,k] * W[e,k] + bias[e]
// Epilogue scatters into Q[b,h,n,hd], K[b,h,n,hd], Vt[b,h,hd,n] (bf16).
// ---------------------------------------------------------------------------
__global__ __launch_bounds__(128) void qkv_gemm_kernel(
    const bf16* __restrict__ xb, const bf16* __restrict__ wb,
    const float* __restrict__ bias, bf16* __restrict__ Qb,
    bf16* __restrict__ Kb, bf16* __restrict__ Vtb) {
  __shared__ bf16 Asm[2][64][32];
  __shared__ bf16 Wsm[2][64][32];

  const int tid = threadIdx.x, wid = tid >> 5, lane = tid & 31;
  const int g = lane >> 4, cl = lane & 15;
  const int et = blockIdx.x % (E3_ / 64);
  const int mt = blockIdx.x / (E3_ / 64);
  const int m0blk = mt * 64;
  const int m0 = m0blk + wid * 16;
  const int e0 = et * 64;

  const v8f zero = {0.f, 0.f, 0.f, 0.f, 0.f, 0.f, 0.f, 0.f};
  v8f acc[4] = {zero, zero, zero, zero};

  gemm_tile_pipeline(xb + (size_t)m0blk * D_, wb + (size_t)e0 * D_,
                     Asm, Wsm, tid, wid, g, cl, acc);

#pragma unroll
  for (int t = 0; t < 4; ++t) {
    const int e = e0 + t * 16 + cl;
    const float bv = bias[e];
    const int which = e / D_;           // 0=q, 1=k, 2=v
    const int dcol = e % D_;
    const int h = dcol >> 6, hi = dcol & 63;
#pragma unroll
    for (int j = 0; j < 8; ++j) {
      const int m = m0 + j + 8 * g;     // C layout: row = vgpr + 8*(lane/16)
      const int b = m / N_, n = m % N_;
      const bf16 val = (bf16)(acc[t][j] + bv);
      if (which == 0)
        Qb[((size_t)(b * H_ + h) * NPAD + n) * HD_ + hi] = val;
      else if (which == 1)
        Kb[((size_t)(b * H_ + h) * NPAD + n) * HD_ + hi] = val;
      else
        Vtb[((size_t)(b * H_ + h) * HD_ + hi) * NPAD + n] = val;
    }
  }
}

// ---------------------------------------------------------------------------
// Kernel 3: flash attention with scale-causal mask.
// One wave per 16-row q tile; kv in steps of 32 (4 WMMA for S + 4 for PV).
// ---------------------------------------------------------------------------
__global__ __launch_bounds__(128) void attn_kernel(
    const bf16* __restrict__ Qb, const bf16* __restrict__ Kb,
    const bf16* __restrict__ Vtb, bf16* __restrict__ Ob) {
  __shared__ bf16 Pls[4][16][32];       // per-wave P staging (C-layout -> A-frag)

  const int tid = threadIdx.x, wid = tid >> 5, lane = tid & 31;
  const int g = lane >> 4, cl = lane & 15;
  const int qtg = blockIdx.x % 6;
  const int bh = blockIdx.x / 6;
  const int h = bh % H_, b = bh / H_;
  const int qt = qtg * 4 + wid;
  if (qt >= (N_ + 15) / 16) return;     // 22 q tiles
  const int q0 = qt * 16;

  const bf16* Qbase = Qb + (size_t)(b * H_ + h) * NPAD * HD_;
  const bf16* Kbase = Kb + (size_t)(b * H_ + h) * NPAD * HD_;
  const bf16* Vbase = Vtb + (size_t)(b * H_ + h) * HD_ * NPAD;

  const bf16* qrow = Qbase + (size_t)(q0 + cl) * HD_;
  const v16bf qf0 = load_frag(qrow, 0, g);
  const v16bf qf1 = load_frag(qrow, 32, g);

  const v8f zero = {0.f, 0.f, 0.f, 0.f, 0.f, 0.f, 0.f, 0.f};
  v8f o[4] = {zero, zero, zero, zero};
  float m8[8], l8[8];
#pragma unroll
  for (int j = 0; j < 8; ++j) { m8[j] = -__builtin_inff(); l8[j] = 0.f; }

  bf16* Pw = &Pls[wid][0][0];

  for (int kv0 = 0; kv0 < NPAD; kv0 += 32) {
    // ---- S tile (16 q x 32 kv), hd=64 split into two K=32 WMMA chunks ----
    v8f s0 = zero, s1 = zero;
    {
      const bf16* krow0 = Kbase + (size_t)(kv0 + cl) * HD_;
      __builtin_prefetch(krow0 + 32 * HD_, 0, 0);
      s0 = WMMA_BF16(qf0, load_frag(krow0, 0, g), s0);
      s0 = WMMA_BF16(qf1, load_frag(krow0, 32, g), s0);
      const bf16* krow1 = Kbase + (size_t)(kv0 + 16 + cl) * HD_;
      s1 = WMMA_BF16(qf0, load_frag(krow1, 0, g), s1);
      s1 = WMMA_BF16(qf1, load_frag(krow1, 32, g), s1);
    }

    const int k0i = kv0 + cl, k1i = kv0 + 16 + cl;
    const int sk0 = scale_of(k0i < N_ ? k0i : 0);
    const int sk1 = scale_of(k1i < N_ ? k1i : 0);
    const bool k0ok = (k0i < N_), k1ok = (k1i < N_);

#pragma unroll
    for (int j = 0; j < 8; ++j) {
      const int q = q0 + j + 8 * g;
      const int sq = scale_of(q < N_ ? q : N_ - 1);
      const bool qall = (q == 0);       // CLS row attends everywhere
      float v0 = (k0ok && (qall || sk0 <= sq)) ? s0[j] * 0.125f : -__builtin_inff();
      float v1 = (k1ok && (qall || sk1 <= sq)) ? s1[j] * 0.125f : -__builtin_inff();
      // row max over the 16 lanes of this half-wave (C layout keeps g fixed)
      float rmax = fmaxf(v0, v1);
      rmax = fmaxf(rmax, __shfl_xor(rmax, 1, 32));
      rmax = fmaxf(rmax, __shfl_xor(rmax, 2, 32));
      rmax = fmaxf(rmax, __shfl_xor(rmax, 4, 32));
      rmax = fmaxf(rmax, __shfl_xor(rmax, 8, 32));
      const float nm = fmaxf(m8[j], rmax);
      const float corr = __expf(m8[j] - nm);
      const float p0 = __expf(v0 - nm);
      const float p1 = __expf(v1 - nm);
      float rs = p0 + p1;
      rs += __shfl_xor(rs, 1, 32);
      rs += __shfl_xor(rs, 2, 32);
      rs += __shfl_xor(rs, 4, 32);
      rs += __shfl_xor(rs, 8, 32);
      l8[j] = l8[j] * corr + rs;
      m8[j] = nm;
      o[0][j] *= corr; o[1][j] *= corr; o[2][j] *= corr; o[3][j] *= corr;
      const int prow = j + 8 * g;
      Pw[prow * 32 + cl] = (bf16)p0;
      Pw[prow * 32 + 16 + cl] = (bf16)p1;
    }

    // wave-local LDS RAW: P C-layout stores -> A-fragment loads
    asm volatile("s_wait_dscnt 0" ::: "memory");
    const v16bf pf = load_frag(Pw + cl * 32, 0, g);

    // ---- O += P @ V : Vt is [hd, kv], so each lane reads contiguous kv ----
#pragma unroll
    for (int t = 0; t < 4; ++t) {
      const bf16* vrow = Vbase + (size_t)(t * 16 + cl) * NPAD + kv0;
      o[t] = WMMA_BF16(pf, load_frag(vrow, 0, g), o[t]);
    }
  }

#pragma unroll
  for (int t = 0; t < 4; ++t) {
#pragma unroll
    for (int j = 0; j < 8; ++j) {
      const int q = q0 + j + 8 * g;
      if (q < N_)
        Ob[((size_t)(b * N_ + q)) * D_ + h * HD_ + t * 16 + cl] =
            (bf16)(o[t][j] / l8[j]);
    }
  }
}

// ---------------------------------------------------------------------------
// Kernel 4: output projection. out[m,e] = sum_k A[m,k] * Wp[e,k] + bias[e]
// Writes f32 directly to d_out.
// ---------------------------------------------------------------------------
__global__ __launch_bounds__(128) void proj_gemm_kernel(
    const bf16* __restrict__ ab, const bf16* __restrict__ wb,
    const float* __restrict__ bias, float* __restrict__ out) {
  __shared__ bf16 Asm[2][64][32];
  __shared__ bf16 Wsm[2][64][32];

  const int tid = threadIdx.x, wid = tid >> 5, lane = tid & 31;
  const int g = lane >> 4, cl = lane & 15;
  const int et = blockIdx.x % (D_ / 64);
  const int mt = blockIdx.x / (D_ / 64);
  const int m0blk = mt * 64;
  const int m0 = m0blk + wid * 16;
  const int e0 = et * 64;

  const v8f zero = {0.f, 0.f, 0.f, 0.f, 0.f, 0.f, 0.f, 0.f};
  v8f acc[4] = {zero, zero, zero, zero};

  gemm_tile_pipeline(ab + (size_t)m0blk * D_, wb + (size_t)e0 * D_,
                     Asm, Wsm, tid, wid, g, cl, acc);

#pragma unroll
  for (int t = 0; t < 4; ++t) {
    const int e = e0 + t * 16 + cl;
    const float bv = bias[e];
#pragma unroll
    for (int j = 0; j < 8; ++j) {
      const int m = m0 + j + 8 * g;
      out[(size_t)m * D_ + e] = acc[t][j] + bv;
    }
  }
}

// ---------------------------------------------------------------------------
// Host-side orchestration.
// ---------------------------------------------------------------------------
extern "C" void kernel_launch(void* const* d_in, const int* in_sizes, int n_in,
                              void* d_out, int out_size, void* d_ws, size_t ws_size,
                              hipStream_t stream) {
  const float* x      = (const float*)d_in[0];  // [64, 341, 768]
  const float* qkv_w  = (const float*)d_in[1];  // [2304, 768]
  const float* qkv_b  = (const float*)d_in[2];  // [2304]
  const float* proj_w = (const float*)d_in[3];  // [768, 768]
  const float* proj_b = (const float*)d_in[4];  // [768]
  float* out = (float*)d_out;                   // [64, 341, 768]

  // Workspace partition (bytes; every block is 256B-aligned by construction)
  char* ws = (char*)d_ws;
  const size_t XB_BYTES   = (size_t)M_ * D_ * sizeof(bf16);       // 33,521,664
  const size_t QKVW_BYTES = (size_t)E3_ * D_ * sizeof(bf16);      //  3,538,944
  const size_t PW_BYTES   = (size_t)D_ * D_ * sizeof(bf16);       //  1,179,648
  const size_t HEAD_BYTES = (size_t)B_ * H_ * NPAD * HD_ * sizeof(bf16); // 34,603,008

  size_t off = 0;
  bf16* xb   = (bf16*)(ws + off); off += XB_BYTES;
  bf16* wqkv = (bf16*)(ws + off); off += QKVW_BYTES;
  bf16* wprj = (bf16*)(ws + off); off += PW_BYTES;
  bf16* Qb   = (bf16*)(ws + off); off += HEAD_BYTES;
  bf16* Kb   = (bf16*)(ws + off); off += HEAD_BYTES;
  bf16* Vtb  = (bf16*)(ws + off); off += HEAD_BYTES;
  bf16* Ob   = (bf16*)(ws + off); off += XB_BYTES;
  (void)ws_size; (void)in_sizes; (void)n_in; (void)out_size;

  // 1) downcast f32 -> bf16
  cvt_f32_bf16_kernel<<<2048, 256, 0, stream>>>(x, xb, M_ * D_);
  cvt_f32_bf16_kernel<<<1024, 256, 0, stream>>>(qkv_w, wqkv, E3_ * D_);
  cvt_f32_bf16_kernel<<<512, 256, 0, stream>>>(proj_w, wprj, D_ * D_);

  // 2) QKV projection: (21824 x 768) @ (2304 x 768)^T
  qkv_gemm_kernel<<<dim3((M_ / 64) * (E3_ / 64)), 128, 0, stream>>>(
      xb, wqkv, qkv_b, Qb, Kb, Vtb);

  // 3) flash attention: B*H*ceil(22/4) blocks, 4 waves each
  attn_kernel<<<dim3(B_ * H_ * 6), 128, 0, stream>>>(Qb, Kb, Vtb, Ob);

  // 4) output projection: (21824 x 768) @ (768 x 768)^T -> f32 out
  proj_gemm_kernel<<<dim3((M_ / 64) * (D_ / 64)), 128, 0, stream>>>(
      Ob, wprj, proj_b, out);
}
